// GCRNN_HGP_87342454931747
// MI455X (gfx1250) — matrix operand order repaired and checked by reading, not compiled, not run on hardware
//
#include <hip/hip_runtime.h>
#include <hip/hip_bf16.h>
#include <math.h>

#define NNODES 50000
#define NEDGES 1600000
#define INC    128
#define HIDC   128
#define G3     384
#define TOPK   25000
#define SORTM  65536

typedef __bf16 bf16_t;
typedef __attribute__((ext_vector_type(16))) __bf16 v16bf;
typedef __attribute__((ext_vector_type(8)))  __bf16 v8bf;
typedef __attribute__((ext_vector_type(8)))  float  v8f;

// ---------------- small elementwise kernels ----------------

__global__ void init_deg_k(float* deg, int n) {
    int i = blockIdx.x * blockDim.x + threadIdx.x;
    if (i < n) deg[i] = 1.0f;                    // self loop
}

__global__ void accum_deg_k(const int* __restrict__ dst, float* deg, int e) {
    int i = blockIdx.x * blockDim.x + threadIdx.x;
    if (i < e) atomicAdd(&deg[dst[i]], 1.0f);
}

__global__ void dinv_k(float* deg, float* selfc, int n) {
    int i = blockIdx.x * blockDim.x + threadIdx.x;
    if (i < n) {
        float d = rsqrtf(deg[i]);
        deg[i] = d;                              // deg buffer becomes dinv
        selfc[i] = d * d;
    }
}

__global__ void coef_k(const int* __restrict__ src, const int* __restrict__ dst,
                       const float* __restrict__ dinv, float* coef, int e) {
    int i = blockIdx.x * blockDim.x + threadIdx.x;
    if (i < e) coef[i] = dinv[src[i]] * dinv[dst[i]];
}

__global__ void zero_k(float* p, long n) {
    long i = (long)blockIdx.x * blockDim.x + threadIdx.x;
    if (i < n) p[i] = 0.0f;
}

// convert fp32 activations to row-major bf16
__global__ void cvt_bf16_k(const float* __restrict__ in, bf16_t* out, long n) {
    long i = (long)blockIdx.x * blockDim.x + threadIdx.x;
    if (i < n) out[i] = (bf16_t)in[i];
}

// Pack a weight matrix into the WMMA B-fragment register layout:
// fragment (t = column tile, kk = k-step of 32) -> 32 lanes x 16 bf16 contiguous.
// transposed=0: source W[k*ncol + n] (k-major, e.g. W1 [128,128])
// transposed=1: source W[n*128  + k] (row-major weight, e.g. W_ih [384,128])
__global__ void pack_b_k(const float* __restrict__ W, bf16_t* out, int ncol, int transposed) {
    int id = blockIdx.x * blockDim.x + threadIdx.x;
    int total = ncol * 128;                      // ntiles*4*32*16 = ncol*128
    if (id >= total) return;
    int i  = id & 15;
    int l  = (id >> 4) & 31;
    int kk = (id >> 9) & 3;
    int t  = id >> 11;
    int k  = kk * 32 + ((l >= 16) ? 8 : 0) + (i & 7) + ((i >= 8) ? 16 : 0);
    int n  = t * 16 + (l & 15);
    float v = transposed ? W[n * INC + k] : W[k * ncol + n];
    out[id] = (bf16_t)v;
}

// ---------------- WMMA bf16 GEMM: C = A(bf16,row-major) * Bpacked + bias ----------
// one wave computes a 16-row strip across all column tiles; A resident in regs.
// nrows must be a multiple of 16 (true here: 50000 = 3125*16), so no row guards.

__global__ void wmma_gemm_k(const bf16_t* __restrict__ A, const bf16_t* __restrict__ Bp,
                            const float* __restrict__ bias, float* __restrict__ C,
                            int nrows, int ncol) {
    int wave = threadIdx.x >> 5;
    int lane = threadIdx.x & 31;
    int row0 = (blockIdx.x * 8 + wave) * 16;
    if (row0 >= nrows) return;                   // uniform per wave: EXEC stays all-1s

    int  m  = lane & 15;
    bool hi = lane >= 16;
    const bf16_t* arow = A + (size_t)(row0 + m) * INC;

    // A fragments for 4 k-steps of 32 (ISA 16-bit A 16x32 layout): two 16B vector loads each
    v16bf a[4];
#pragma unroll
    for (int kk = 0; kk < 4; ++kk) {
        int k0 = kk * 32 + (hi ? 8 : 0);
        v8bf lo  = *(const v8bf*)(arow + k0);
        v8bf hi8 = *(const v8bf*)(arow + k0 + 16);
        a[kk] = __builtin_shufflevector(lo, hi8, 0, 1, 2, 3, 4, 5, 6, 7,
                                                 8, 9, 10, 11, 12, 13, 14, 15);
    }

    int ntiles = ncol >> 4;
    int ncolb  = lane & 15;
    int rbase  = row0 + (hi ? 8 : 0);
    for (int t = 0; t < ntiles; ++t) {
        int n = t * 16 + ncolb;
        v8f acc = {0.f, 0.f, 0.f, 0.f, 0.f, 0.f, 0.f, 0.f};
#pragma unroll
        for (int kk = 0; kk < 4; ++kk) {
            // pre-packed fragment: one aligned 32B load per lane
            v16bf b = *(const v16bf*)(Bp + (((size_t)t * 4 + kk) * 32 + lane) * 16);
            acc = __builtin_amdgcn_wmma_f32_16x16x32_bf16(
                false, a[kk], false, b, (short)0, acc, false, false);
        }
        float bv = bias[n];
        float* crow = C + (size_t)rbase * ncol + n;
#pragma unroll
        for (int v = 0; v < 8; ++v)
            crow[(size_t)v * ncol] = acc[v] + bv;   // D layout: VGPR v -> row rbase+v
    }
}

// ---------------- edge scatter: out[dst] += X[src]*coef (one wave per edge) --------

__global__ void scatter_k(const float* __restrict__ X, const int* __restrict__ src,
                          const int* __restrict__ dst, const float* __restrict__ coef,
                          float* out, int e) {
    int g = blockIdx.x * blockDim.x + threadIdx.x;
    int edge = g >> 5;
    if (edge >= e) return;
    int lane = g & 31;
    int s = src[edge], d = dst[edge];
    float c = coef[edge];
    float4 v = ((const float4*)(X + (size_t)s * INC))[lane];
    float* o = out + (size_t)d * INC + lane * 4;
    atomicAdd(o + 0, v.x * c);
    atomicAdd(o + 1, v.y * c);
    atomicAdd(o + 2, v.z * c);
    atomicAdd(o + 3, v.w * c);
}

// h = relu(prop + xw*selfc + b1), written directly as bf16 (only the 2nd GEMM reads h)
__global__ void relu_bias_k(const float* __restrict__ prop, const float* __restrict__ xw,
                            const float* __restrict__ selfc, const float* __restrict__ b1,
                            bf16_t* h, long total) {
    long i = (long)blockIdx.x * blockDim.x + threadIdx.x;
    if (i >= total) return;
    long n = i >> 7;
    int  c = (int)(i & 127);
    float v = prop[i] + xw[i] * selfc[n] + b1[c];
    h[i] = (bf16_t)fmaxf(v, 0.0f);
}

// ---------------- GRU scan: one persistent workgroup, W_hh in LDS ----------------

__global__ void gru_scan_k(const float* __restrict__ gi, const float* __restrict__ Whh,
                           const float* __restrict__ bhh, float* __restrict__ hs, int n) {
    extern __shared__ float smem[];
    float* Whht = smem;                  // [128][384]  Whht[k*384+j] = Whh[j*128+k]
    float* ghs  = smem + 128 * G3;       // [384]
    float* hp   = ghs  + G3;             // [128]
    int j = threadIdx.x;                 // 384 threads

    for (int idx = j; idx < G3 * HIDC; idx += G3) {
        int jj = idx / HIDC, k = idx % HIDC;
        Whht[k * G3 + jj] = Whh[idx];
    }
    if (j < HIDC) hp[j] = 0.0f;
    float bh = bhh[j];
    __syncthreads();

    for (int t = 0; t < n; ++t) {
        float gh = bh;
#pragma unroll 8
        for (int k = 0; k < HIDC; ++k)
            gh += Whht[k * G3 + j] * hp[k];
        ghs[j] = gh;
        __syncthreads();
        if (j < HIDC) {
            const float* g = gi + (size_t)t * G3;
            float rg = 1.0f / (1.0f + expf(-(g[j]       + ghs[j])));
            float zg = 1.0f / (1.0f + expf(-(g[128 + j] + ghs[128 + j])));
            float ng = tanhf(g[256 + j] + rg * ghs[256 + j]);
            float hn = (1.0f - zg) * ng + zg * hp[j];
            hs[(size_t)t * HIDC + j] = hn;
            hp[j] = hn;
        }
        __syncthreads();
    }
}

// ---------------- score = || hs - propagate(hs) ||_1 (wave per node) ----------------

__global__ void score_k(const float* __restrict__ hs, const float* __restrict__ prop,
                        const float* __restrict__ selfc, float* score, int n) {
    int g = blockIdx.x * blockDim.x + threadIdx.x;
    int node = g >> 5;
    if (node >= n) return;
    int lane = g & 31;
    float4 hv = ((const float4*)(hs   + (size_t)node * INC))[lane];
    float4 pv = ((const float4*)(prop + (size_t)node * INC))[lane];
    float sc = selfc[node];
    float s = fabsf(hv.x - (pv.x + hv.x * sc)) + fabsf(hv.y - (pv.y + hv.y * sc))
            + fabsf(hv.z - (pv.z + hv.z * sc)) + fabsf(hv.w - (pv.w + hv.w * sc));
#pragma unroll
    for (int off = 16; off; off >>= 1) s += __shfl_down(s, off, 32);
    if (lane == 0) score[node] = s;
}

// ---------------- top-k: composite keys + global bitonic sort ----------------

__global__ void keys_k(const float* __restrict__ score, unsigned long long* keys, int n) {
    int i = blockIdx.x * blockDim.x + threadIdx.x;
    if (i >= SORTM) return;
    unsigned long long key = 0ull;
    if (i < n) {
        unsigned u = __float_as_uint(score[i]);
        unsigned mapped = (u & 0x80000000u) ? ~u : (u | 0x80000000u);   // orderable
        key = ((unsigned long long)mapped << 32) | (unsigned long long)(0xFFFFFFFFu - (unsigned)i);
    }
    keys[i] = key;
}

__global__ void bitonic_k(unsigned long long* keys, int jj, int kk) {
    int i = blockIdx.x * blockDim.x + threadIdx.x;
    int ixj = i ^ jj;
    if (ixj > i) {
        unsigned long long a = keys[i], b = keys[ixj];
        bool up = ((i & kk) == 0);
        bool sw = up ? (a < b) : (a > b);        // descending overall
        if (sw) { keys[i] = b; keys[ixj] = a; }
    }
}

__global__ void perm_k(const unsigned long long* __restrict__ keys, int* perm, int k) {
    int i = blockIdx.x * blockDim.x + threadIdx.x;
    if (i < k) perm[i] = (int)(0xFFFFFFFFu - (unsigned)(keys[i] & 0xFFFFFFFFull));
}

// ---------------- classifier + log_softmax (wave per pooled row) ----------------

__global__ void classify_k(const float* __restrict__ hs, const int* __restrict__ perm,
                           const float* __restrict__ Wfc, const float* __restrict__ bfc,
                           float* out, int k) {
    int g = blockIdx.x * blockDim.x + threadIdx.x;
    int row = g >> 5;
    if (row >= k) return;
    int lane = g & 31;
    int node = perm[row];
    float4 v  = ((const float4*)(hs + (size_t)node * INC))[lane];
    float4 w0 = ((const float4*)Wfc)[lane];
    float4 w1 = ((const float4*)Wfc)[32 + lane];
    float d0 = v.x * w0.x + v.y * w0.y + v.z * w0.z + v.w * w0.w;
    float d1 = v.x * w1.x + v.y * w1.y + v.z * w1.z + v.w * w1.w;
#pragma unroll
    for (int off = 16; off; off >>= 1) {
        d0 += __shfl_down(d0, off, 32);
        d1 += __shfl_down(d1, off, 32);
    }
    if (lane == 0) {
        float l0 = d0 + bfc[0], l1 = d1 + bfc[1];
        float m = fmaxf(l0, l1);
        float lse = m + logf(expf(l0 - m) + expf(l1 - m));
        out[2 * row]     = l0 - lse;
        out[2 * row + 1] = l1 - lse;
    }
}

// ---------------- launch ----------------

extern "C" void kernel_launch(void* const* d_in, const int* in_sizes, int n_in,
                              void* d_out, int out_size, void* d_ws, size_t ws_size,
                              hipStream_t stream) {
    const float* x    = (const float*)d_in[0];
    const int*   ei   = (const int*)d_in[1];
    const float* W1   = (const float*)d_in[2];
    const float* b1   = (const float*)d_in[3];
    const float* Wih  = (const float*)d_in[4];
    const float* Whh  = (const float*)d_in[5];
    const float* bih  = (const float*)d_in[6];
    const float* bhh  = (const float*)d_in[7];
    const float* Wfc  = (const float*)d_in[8];
    const float* bfc  = (const float*)d_in[9];
    float* out = (float*)d_out;

    const int* src = ei;
    const int* dst = ei + NEDGES;

    // carve workspace
    char* ws = (char*)d_ws;
    size_t off = 0;
    auto carve = [&](size_t bytes) -> char* {
        char* p = ws + off;
        off = (off + bytes + 255) & ~(size_t)255;
        return p;
    };
    float*  deg   = (float*)carve((size_t)NNODES * 4);          // then dinv
    float*  selfc = (float*)carve((size_t)NNODES * 4);
    float*  coef  = (float*)carve((size_t)NEDGES * 4);
    bf16_t* w1p   = (bf16_t*)carve((size_t)INC * HIDC * 2);     // packed B fragments
    bf16_t* wihp  = (bf16_t*)carve((size_t)INC * G3 * 2);       // packed B fragments
    float*  zbias = (float*)carve((size_t)HIDC * 4);            // zero bias for GEMM1
    bf16_t* xbf   = (bf16_t*)carve((size_t)NNODES * INC * 2);
    float*  xw    = (float*)carve((size_t)NNODES * HIDC * 4);
    bf16_t* hbf   = (bf16_t*)carve((size_t)NNODES * HIDC * 2);
    float*  gi    = (float*)carve((size_t)NNODES * G3 * 4);
    float*  hs    = (float*)carve((size_t)NNODES * HIDC * 4);
    float*  prop  = (float*)carve((size_t)NNODES * HIDC * 4);
    float*  score = (float*)carve((size_t)NNODES * 4);
    unsigned long long* keys = (unsigned long long*)carve((size_t)SORTM * 8);
    int*    perm  = (int*)carve((size_t)TOPK * 4);
    (void)ws_size; (void)in_sizes; (void)n_in; (void)out_size;

    const int T = 256;
    long tot = (long)NNODES * HIDC;

    // GCN normalization
    init_deg_k <<<(NNODES + T - 1) / T, T, 0, stream>>>(deg, NNODES);
    accum_deg_k<<<(NEDGES + T - 1) / T, T, 0, stream>>>(dst, deg, NEDGES);
    dinv_k     <<<(NNODES + T - 1) / T, T, 0, stream>>>(deg, selfc, NNODES);
    coef_k     <<<(NEDGES + T - 1) / T, T, 0, stream>>>(src, dst, deg, coef, NEDGES);

    // weight packing + activation conversion
    pack_b_k   <<<(INC * HIDC + T - 1) / T, T, 0, stream>>>(W1, w1p, HIDC, 0);
    pack_b_k   <<<(INC * G3 + T - 1) / T, T, 0, stream>>>(Wih, wihp, G3, 1);
    zero_k     <<<1, T, 0, stream>>>(zbias, HIDC);
    cvt_bf16_k <<<(unsigned)((tot + T - 1) / T), T, 0, stream>>>(x, xbf, tot);

    // GCNConv: xw = x @ W1 (WMMA bf16, fp32 accum)
    wmma_gemm_k<<<(NNODES + 127) / 128, T, 0, stream>>>(xbf, w1p, zbias, xw, NNODES, HIDC);
    // propagate + relu (h stored as bf16 for the next GEMM)
    zero_k     <<<(unsigned)((tot + T - 1) / T), T, 0, stream>>>(prop, tot);
    scatter_k  <<<(NEDGES * 32 + T - 1) / T, T, 0, stream>>>(xw, src, dst, coef, prop, NEDGES);
    relu_bias_k<<<(unsigned)((tot + T - 1) / T), T, 0, stream>>>(prop, xw, selfc, b1, hbf, tot);

    // GRU input gates: gi = h @ W_ih.T + b_ih (WMMA bf16)
    wmma_gemm_k<<<(NNODES + 127) / 128, T, 0, stream>>>(hbf, wihp, bih, gi, NNODES, G3);
    // serial GRU scan, W_hh staged in 196KB LDS (CDNA5 320KB/WGP)
    size_t smem = (size_t)(HIDC * G3 + G3 + HIDC) * 4;
    gru_scan_k<<<1, G3, smem, stream>>>(gi, Whh, bhh, hs, NNODES);

    // pooling score
    zero_k     <<<(unsigned)((tot + T - 1) / T), T, 0, stream>>>(prop, tot);
    scatter_k  <<<(NEDGES * 32 + T - 1) / T, T, 0, stream>>>(hs, src, dst, coef, prop, NEDGES);
    score_k    <<<(NNODES * 32 + T - 1) / T, T, 0, stream>>>(hs, prop, selfc, score, NNODES);

    // exact ordered top-k via bitonic sort of composite keys
    keys_k<<<SORTM / T, T, 0, stream>>>(score, keys, NNODES);
    for (int kk = 2; kk <= SORTM; kk <<= 1)
        for (int jj = kk >> 1; jj > 0; jj >>= 1)
            bitonic_k<<<SORTM / T, T, 0, stream>>>(keys, jj, kk);
    perm_k<<<(TOPK + T - 1) / T, T, 0, stream>>>(keys, perm, TOPK);

    // classifier + log_softmax
    classify_k<<<(TOPK * 32 + T - 1) / T, T, 0, stream>>>(hs, perm, Wfc, bfc, out, TOPK);
}
